// BoSsDecoderLayer_22385369547247
// MI455X (gfx1250) — matrix operand order repaired and checked by reading, not compiled, not run on hardware
//
#include <hip/hip_runtime.h>

// ---------------------------------------------------------------------------
// MI455X (gfx1250) BoSs decoder layer.
// Weight-BW-bound (~251MB fp32 weights @ 23.3TB/s ~ 11us floor).
// All matmuls via v_wmma_f32_16x16x32_bf16; weights converted fp32->bf16 with
// hardware cvt (v_cvt_pk_bf16_f32) while streaming; activations kept bf16;
// flash attention per-wave with LDS C->A relayout of P.
// ---------------------------------------------------------------------------

typedef unsigned short u16;
typedef __attribute__((ext_vector_type(16))) __bf16 v16bf;
typedef __attribute__((ext_vector_type(8)))  float  v8f;

struct __align__(16) U4 { unsigned x, y, z, w; };
struct __align__(16) F4 { float x, y, z, w; };

union BF16x16 {
    v16bf v;
    u16   u[16];
    U4    q[2];
};

// fp32 -> bf16 via hardware convert (RNE); lowers to v_cvt_pk_bf16_f32 pairs.
__device__ inline u16 f2bf(float f) {
    union { __bf16 b; u16 u; } cv;
    cv.b = (__bf16)f;
    return cv.u;
}

__device__ inline BF16x16 cvt16(const float* __restrict__ p) {
    const F4* p4 = (const F4*)p;
    F4 a = p4[0], b = p4[1], c = p4[2], d = p4[3];
    float f[16] = { a.x,a.y,a.z,a.w, b.x,b.y,b.z,b.w,
                    c.x,c.y,c.z,c.w, d.x,d.y,d.z,d.w };
    BF16x16 r;
#pragma unroll
    for (int i = 0; i < 16; ++i) r.v[i] = (__bf16)f[i];
    return r;
}

__device__ inline v8f wmma_bf16(v16bf a, v16bf b, v8f c) {
    return __builtin_amdgcn_wmma_f32_16x16x32_bf16(
        /*neg_a=*/false, a, /*neg_b=*/false, b,
        /*c_mod=*/(short)0, c, /*reuse_a=*/false, /*reuse_b=*/false);
}

// ---------------------------------------------------------------------------
// RMSNorm fp32 -> bf16 (one block per row of 2048)
// ---------------------------------------------------------------------------
__global__ __launch_bounds__(256) void rmsnorm_bf16_kernel(
    const float* __restrict__ x, const float* __restrict__ w, u16* __restrict__ out) {
    const int H = 2048;
    int row = blockIdx.x, tid = threadIdx.x;
    const float* xr = x + (size_t)row * H;
    float s = 0.f;
    for (int i = tid; i < H; i += 256) { float v = xr[i]; s += v * v; }
    __shared__ float red[256];
    red[tid] = s; __syncthreads();
    for (int off = 128; off > 0; off >>= 1) {
        if (tid < off) red[tid] += red[tid + off];
        __syncthreads();
    }
    float rstd = rsqrtf(red[0] * (1.0f / H) + 1e-6f);
    for (int i = tid; i < H; i += 256)
        out[(size_t)row * H + i] = f2bf(xr[i] * rstd * w[i]);
}

// ---------------------------------------------------------------------------
// GEMM: C[M,N] = A[M,K](bf16) x W[N,K](fp32->bf16)^T, per-wave 16x16 tile.
// Block = 256 threads = 8 waves covering 16 rows x 128 cols.
// Epilogues: plain f32 store | +residual f32 | silu(g)*u -> bf16 (two weights)
// ---------------------------------------------------------------------------
enum GEpi { EPI_F32 = 0, EPI_RESID_F32 = 1, EPI_SILU2_BF16 = 2 };

template <int E>
__global__ __launch_bounds__(256) void gemm_bf16w32_kernel(
    const u16* __restrict__ A, const float* __restrict__ W0,
    const float* __restrict__ W1, const float* __restrict__ resid,
    void* __restrict__ Cout, int N, int K) {
    int wave = threadIdx.x >> 5, lane = threadIdx.x & 31;
    int l15 = lane & 15, half = lane >> 4;
    int tileM = blockIdx.y * 16;
    int tileN = blockIdx.x * 128 + wave * 16;

    const u16*   arow  = A  + (size_t)(tileM + l15) * K;   // A lane = M row
    const float* w0row = W0 + (size_t)(tileN + l15) * K;   // B lane = N row
    const float* w1row = (E == EPI_SILU2_BF16) ? (W1 + (size_t)(tileN + l15) * K) : nullptr;

    v8f acc0 = {}; v8f acc1 = {};
#pragma unroll 2
    for (int k0 = 0; k0 < K; k0 += 32) {
        __builtin_prefetch(w0row + k0 + 256, 0, 1);
        BF16x16 af;   // A 16x32: lane run K=16h..16h+15
        af.q[0] = *(const U4*)(arow + k0 + 16 * half);
        af.q[1] = *(const U4*)(arow + k0 + 16 * half + 8);
        BF16x16 b0 = cvt16(w0row + k0 + 16 * half);  // B 32x16: lane run K=16h..16h+15
        acc0 = wmma_bf16(af.v, b0.v, acc0);
        if (E == EPI_SILU2_BF16) {
            __builtin_prefetch(w1row + k0 + 256, 0, 1);
            BF16x16 b1 = cvt16(w1row + k0 + 16 * half);
            acc1 = wmma_bf16(af.v, b1.v, acc1);
        }
    }
    int col = tileN + l15;
#pragma unroll
    for (int r = 0; r < 8; ++r) {
        int row = tileM + r + 8 * half;          // C layout: M = r + 8*(lane>=16)
        size_t idx = (size_t)row * N + col;
        if (E == EPI_F32) {
            ((float*)Cout)[idx] = acc0[r];
        } else if (E == EPI_RESID_F32) {
            ((float*)Cout)[idx] = acc0[r] + resid[idx];
        } else {
            float g = acc0[r], u = acc1[r];
            ((u16*)Cout)[idx] = f2bf(g / (1.f + __expf(-g)) * u);
        }
    }
}

// ---------------------------------------------------------------------------
// RoPE fp32 [s, H*128] -> bf16 [H][s][128]; q also absorbs 1/sqrt(d) scale.
// ---------------------------------------------------------------------------
__global__ __launch_bounds__(256) void rope_kernel(
    const float* __restrict__ in, const int* __restrict__ pos,
    u16* __restrict__ out, int nheads, float scale) {
    int t = blockIdx.x * 256 + threadIdx.x;
    if (t >= nheads * 2048 * 64) return;
    int j = t & 63; int rem = t >> 6;
    int spos = rem & 2047; int h = rem >> 11;
    float p = (float)pos[spos];
    // inv_freq[j] = theta^(-2j/128) = exp(-j * (2/128) * ln(10000))
    float ang = p * __expf((float)j * (-2.0f / 128.0f) * 9.210340372f);
    float sn, c; __sincosf(ang, &sn, &c);
    int ldin = nheads * 128;
    float x0 = in[(size_t)spos * ldin + h * 128 + j];
    float x1 = in[(size_t)spos * ldin + h * 128 + j + 64];
    size_t ob = ((size_t)h * 2048 + spos) * 128 + j;
    out[ob]      = f2bf((x0 * c - x1 * sn) * scale);
    out[ob + 64] = f2bf((x1 * c + x0 * sn) * scale);
}

// V fp32 [s, KVH*128] -> bf16 Vt [KVH][128][s]  (for PV B-operand contiguity)
__global__ __launch_bounds__(256) void vtrans_kernel(
    const float* __restrict__ v, u16* __restrict__ vt) {
    int t = blockIdx.x * 256 + threadIdx.x;   // [kvh][d][s] flat
    int spos = t & 2047; int rem = t >> 11;
    int d = rem & 127; int kvh = rem >> 7;
    vt[t] = f2bf(v[(size_t)spos * 1024 + kvh * 128 + d]);
}

// ---------------------------------------------------------------------------
// Flash attention, per-wave 16-row q tile. WINDOW(4096) >= s(2048) so only
// same-sid & causal masking applies. Scale folded into q at RoPE time.
// Q [h][s][128] bf16, K [kvh][s][128] bf16, Vt [kvh][128][s] bf16.
// ---------------------------------------------------------------------------
__global__ __launch_bounds__(128) void attn_kernel(
    const u16* __restrict__ Q, const u16* __restrict__ Kmat,
    const u16* __restrict__ Vt, const int* __restrict__ sid,
    u16* __restrict__ Out) {
    const int S = 2048, D = 128;
    int wave = threadIdx.x >> 5, lane = threadIdx.x & 31;
    int l15 = lane & 15, half = lane >> 4;
    int head = blockIdx.y, kvh = head >> 1;      // GQA rep = 2
    int q0 = (blockIdx.x * 4 + wave) * 16;

    __shared__ u16 plds[4][16 * 32];             // per-wave P staging (C->A layout)
    u16* pl = plds[wave];

    const u16* qrow  = Q    + (((size_t)head * S) + q0 + l15) * D;
    const u16* kbase = Kmat + (size_t)kvh * S * D;
    const u16* vbase = Vt   + (size_t)kvh * D * S;

    BF16x16 qf[4];                               // Q A-frags, dstep = 32*j
#pragma unroll
    for (int j = 0; j < 4; ++j) {
        qf[j].q[0] = *(const U4*)(qrow + 32 * j + 16 * half);
        qf[j].q[1] = *(const U4*)(qrow + 32 * j + 16 * half + 8);
    }
    int qidx[8], sidq[8];
#pragma unroll
    for (int r = 0; r < 8; ++r) { qidx[r] = q0 + r + 8 * half; sidq[r] = sid[qidx[r]]; }

    float mi[8], li[8]; v8f acc[8]; v8f zf = {};
#pragma unroll
    for (int r = 0; r < 8; ++r) { mi[r] = -1e30f; li[r] = 0.f; }
#pragma unroll
    for (int dt = 0; dt < 8; ++dt) acc[dt] = zf;

    for (int kc = 0; kc < q0 + 16; kc += 32) {
        v8f s0 = {}, s1 = {};
        const u16* k0p = kbase + (size_t)(kc + l15) * D + 16 * half;
        const u16* k1p = k0p + (size_t)16 * D;
#pragma unroll
        for (int j = 0; j < 4; ++j) {            // K = d, 4 x 32
            BF16x16 b0, b1;
            b0.q[0] = *(const U4*)(k0p + 32 * j);
            b0.q[1] = *(const U4*)(k0p + 32 * j + 8);
            s0 = wmma_bf16(qf[j].v, b0.v, s0);
            b1.q[0] = *(const U4*)(k1p + 32 * j);
            b1.q[1] = *(const U4*)(k1p + 32 * j + 8);
            s1 = wmma_bf16(qf[j].v, b1.v, s1);
        }
        int sk0 = sid[kc + l15];
        int sk1 = sid[kc + 16 + l15];
        float sc[8];
#pragma unroll
        for (int r = 0; r < 8; ++r) {
            float v0 = ((kc + l15 <= qidx[r]) && (sk0 == sidq[r])) ? s0[r] : -1e30f;
            float v1 = ((kc + 16 + l15 <= qidx[r]) && (sk1 == sidq[r])) ? s1[r] : -1e30f;
            float rm = fmaxf(v0, v1);            // row reduce across 16-lane half
            rm = fmaxf(rm, __shfl_xor(rm, 1, 32));
            rm = fmaxf(rm, __shfl_xor(rm, 2, 32));
            rm = fmaxf(rm, __shfl_xor(rm, 4, 32));
            rm = fmaxf(rm, __shfl_xor(rm, 8, 32));
            float mnew = fmaxf(mi[r], rm);
            float sfac = __expf(mi[r] - mnew);
            float p0 = (v0 > -1e29f) ? __expf(v0 - mnew) : 0.f;
            float p1 = (v1 > -1e29f) ? __expf(v1 - mnew) : 0.f;
            float rs = p0 + p1;
            rs += __shfl_xor(rs, 1, 32);
            rs += __shfl_xor(rs, 2, 32);
            rs += __shfl_xor(rs, 4, 32);
            rs += __shfl_xor(rs, 8, 32);
            li[r] = li[r] * sfac + rs;
            mi[r] = mnew; sc[r] = sfac;
            int mrow = r + 8 * half;             // C layout -> LDS 16x32
            pl[mrow * 32 + l15]      = f2bf(p0);
            pl[mrow * 32 + 16 + l15] = f2bf(p1);
        }
#pragma unroll
        for (int dt = 0; dt < 8; ++dt)
#pragma unroll
            for (int r = 0; r < 8; ++r) acc[dt][r] *= sc[r];

        __builtin_amdgcn_wave_barrier();
        asm volatile("s_wait_dscnt 0" ::: "memory");  // same-wave DS are in-order; fence
        BF16x16 pa;                              // reload P in A layout (16x32 keys)
        pa.q[0] = *(const U4*)(pl + l15 * 32 + 16 * half);
        pa.q[1] = *(const U4*)(pl + l15 * 32 + 16 * half + 8);
        __builtin_amdgcn_wave_barrier();
#pragma unroll
        for (int dt = 0; dt < 8; ++dt) {         // PV: B = Vt rows (d), contiguous s
            const u16* vp = vbase + (size_t)(dt * 16 + l15) * S + kc + 16 * half;
            BF16x16 vb;
            vb.q[0] = *(const U4*)(vp);
            vb.q[1] = *(const U4*)(vp + 8);
            acc[dt] = wmma_bf16(pa.v, vb.v, acc[dt]);
        }
    }
#pragma unroll
    for (int r = 0; r < 8; ++r) {
        float inv = 1.f / li[r];
        int row = q0 + r + 8 * half;
#pragma unroll
        for (int dt = 0; dt < 8; ++dt)
            Out[(size_t)row * 2048 + head * 128 + dt * 16 + l15] = f2bf(acc[dt][r] * inv);
    }
}

// ---------------------------------------------------------------------------
// Host-side orchestration. Workspace phases overlap; peak 64 MB.
// ---------------------------------------------------------------------------
extern "C" void kernel_launch(void* const* d_in, const int* in_sizes, int n_in,
                              void* d_out, int out_size, void* d_ws, size_t ws_size,
                              hipStream_t stream) {
    const float* hidden = (const float*)d_in[0];
    const int*   sid    = (const int*)d_in[1];
    const int*   pos    = (const int*)d_in[2];
    const float* ln1    = (const float*)d_in[3];
    const float* wq     = (const float*)d_in[4];
    const float* wk     = (const float*)d_in[5];
    const float* wv     = (const float*)d_in[6];
    const float* wo     = (const float*)d_in[7];
    const float* ln2    = (const float*)d_in[8];
    const float* wg     = (const float*)d_in[9];
    const float* wu     = (const float*)d_in[10];
    const float* wd     = (const float*)d_in[11];
    float* out = (float*)d_out;
    char* ws = (char*)d_ws;
    const size_t MB = 1024 * 1024;

    u16*   xn  = (u16*)  (ws + 0);        // [2048,2048] bf16      (phase A)
    float* qf  = (float*)(ws + 8  * MB);  // [2048,2048] f32
    float* kf  = (float*)(ws + 24 * MB);  // [2048,1024] f32
    float* vf  = (float*)(ws + 32 * MB);  // [2048,1024] f32
    u16*   qb  = (u16*)  (ws + 40 * MB);  // [16][2048][128] bf16
    u16*   kb  = (u16*)  (ws + 48 * MB);  // [8][2048][128] bf16
    u16*   vt  = (u16*)  (ws + 52 * MB);  // [8][128][2048] bf16
    u16*   at  = (u16*)  (ws + 8  * MB);  // [2048,2048] bf16      (reuse qf)
    float* h   = (float*)(ws + 16 * MB);  // [2048,2048] f32       (reuse qf/kf)
    u16*   y   = (u16*)  (ws + 0);        // [2048,2048] bf16      (reuse xn)
    u16*   act = (u16*)  (ws + 32 * MB);  // [2048,8192] bf16      (reuse qb/kb/vt)

    rmsnorm_bf16_kernel<<<2048, 256, 0, stream>>>(hidden, ln1, xn);

    gemm_bf16w32_kernel<EPI_F32><<<dim3(16, 128), 256, 0, stream>>>(xn, wq, nullptr, nullptr, qf, 2048, 2048);
    gemm_bf16w32_kernel<EPI_F32><<<dim3(8,  128), 256, 0, stream>>>(xn, wk, nullptr, nullptr, kf, 1024, 2048);
    gemm_bf16w32_kernel<EPI_F32><<<dim3(8,  128), 256, 0, stream>>>(xn, wv, nullptr, nullptr, vf, 1024, 2048);

    rope_kernel<<<16 * 2048 * 64 / 256, 256, 0, stream>>>(qf, pos, qb, 16, 0.08838834764831845f);
    rope_kernel<<<8  * 2048 * 64 / 256, 256, 0, stream>>>(kf, pos, kb, 8, 1.0f);
    vtrans_kernel<<<8 * 128 * 2048 / 256, 256, 0, stream>>>(vf, vt);

    attn_kernel<<<dim3(32, 16), 128, 0, stream>>>(qb, kb, vt, sid, at);

    gemm_bf16w32_kernel<EPI_RESID_F32><<<dim3(16, 128), 256, 0, stream>>>(at, wo, nullptr, hidden, h, 2048, 2048);

    rmsnorm_bf16_kernel<<<2048, 256, 0, stream>>>(h, ln2, y);

    gemm_bf16w32_kernel<EPI_SILU2_BF16><<<dim3(64, 128), 256, 0, stream>>>(y, wg, wu, nullptr, act, 8192, 2048);
    gemm_bf16w32_kernel<EPI_RESID_F32><<<dim3(16, 128), 256, 0, stream>>>(act, wd, nullptr, h, out, 2048, 8192);
}
